// GCN_91319594648342
// MI455X (gfx1250) — matrix operand order repaired and checked by reading, not compiled, not run on hardware
//
#include <hip/hip_runtime.h>
#include <hip/hip_bf16.h>
#include <stdint.h>

// ---------------------------------------------------------------------------
// GCN forward for MI455X (gfx1250, wave32, WMMA).
//   h0     = x @ W1 + b1                       (bf16 WMMA, f32 accum)
//   h      = relu(segment_sum(w * h0[cols]))   (f32 atomics, packed-bf16 gathers)
//   l0     = h @ W2 + b2                       (bf16 WMMA, relu fused in A load)
//   logits = segment_sum(w * l0[cols])
//   out    = log_softmax(logits, axis=1)
// ---------------------------------------------------------------------------

typedef __attribute__((ext_vector_type(16))) __bf16   v16bf;
typedef __attribute__((ext_vector_type(8)))  float    v8f;
typedef __attribute__((ext_vector_type(4)))  float    v4f;
typedef __attribute__((ext_vector_type(4)))  unsigned int v4u;

#define NFEAT  512
#define NHID   256
#define NCLASS 40

union BFrag { v4u q[2]; v16bf v; };   // 32B: one WMMA B fragment per lane

// ---------------------------------------------------------------------------
// Prepack W1 [512,256] f32 -> bf16 WMMA B-fragments.
// Fragment layout (16-bit B, 32x16 KxN, wave32):
//   lane n (0..15)  : VGPR j holds B[kb+2j][nb+n]   , B[kb+2j+1][nb+n]
//   lane n+16       : VGPR j holds B[kb+16+2j][nb+n], B[kb+17+2j][nb+n]
// Packed index: ((kt*NT + nt)*32 + lane)*8 + j, two bf16 per entry.
// ---------------------------------------------------------------------------
__global__ void pack_w1_kernel(const float* __restrict__ W1, __bf16* __restrict__ w1p) {
    int idx = blockIdx.x * 256 + threadIdx.x;          // 16kt * 16nt * 32 * 8 = 65536
    if (idx >= 16 * 16 * 32 * 8) return;
    int j    = idx & 7;
    int lane = (idx >> 3) & 31;
    int nt   = (idx >> 8) & 15;
    int kt   = idx >> 12;
    int hb = lane >> 4, n = lane & 15;
    int k   = kt * 32 + hb * 16 + 2 * j;
    int col = nt * 16 + n;
    w1p[idx * 2 + 0] = (__bf16)W1[(size_t)k       * NHID + col];
    w1p[idx * 2 + 1] = (__bf16)W1[(size_t)(k + 1) * NHID + col];
}

// W2 [256,40] f32 -> bf16 fragments, N zero-padded to 48.
__global__ void pack_w2_kernel(const float* __restrict__ W2, __bf16* __restrict__ w2p) {
    int idx = blockIdx.x * 256 + threadIdx.x;          // 8kt * 3nt * 32 * 8 = 6144
    if (idx >= 8 * 3 * 32 * 8) return;
    int j    = idx & 7;
    int lane = (idx >> 3) & 31;
    int t    = idx >> 8;                               // kt*3 + nt
    int nt = t % 3, kt = t / 3;
    int hb = lane >> 4, n = lane & 15;
    int k   = kt * 32 + hb * 16 + 2 * j;
    int col = nt * 16 + n;
    float v0 = (col < NCLASS) ? W2[(size_t)k       * NCLASS + col] : 0.0f;
    float v1 = (col < NCLASS) ? W2[(size_t)(k + 1) * NCLASS + col] : 0.0f;
    w2p[idx * 2 + 0] = (__bf16)v0;
    w2p[idx * 2 + 1] = (__bf16)v1;
}

// ---------------------------------------------------------------------------
// Build the 16-bit A fragment (16x32 MxK): lane-half h holds
// K in {kb+16h..kb+16h+15} (VGPR 0..7), matching the packed B layout.
// ---------------------------------------------------------------------------
__device__ __forceinline__ v16bf load_a_frag_f32(const float* rowPtr, int kb, int h) {
    const v4f* p0 = (const v4f*)(rowPtr + kb + 16 * h);
    const v4f* p1 = (const v4f*)(rowPtr + kb + 16 * h + 8);
    v4f f0 = p0[0], f1 = p0[1], f2 = p1[0], f3 = p1[1];
    v16bf a;
    a[0]=(__bf16)f0.x; a[1]=(__bf16)f0.y; a[2]=(__bf16)f0.z; a[3]=(__bf16)f0.w;
    a[4]=(__bf16)f1.x; a[5]=(__bf16)f1.y; a[6]=(__bf16)f1.z; a[7]=(__bf16)f1.w;
    a[8]=(__bf16)f2.x; a[9]=(__bf16)f2.y; a[10]=(__bf16)f2.z; a[11]=(__bf16)f2.w;
    a[12]=(__bf16)f3.x; a[13]=(__bf16)f3.y; a[14]=(__bf16)f3.z; a[15]=(__bf16)f3.w;
    return a;
}

__device__ __forceinline__ v16bf load_a_frag_relu(const float* rowPtr, int kb, int h) {
    const v4f* p0 = (const v4f*)(rowPtr + kb + 16 * h);
    const v4f* p1 = (const v4f*)(rowPtr + kb + 16 * h + 8);
    v4f f0 = p0[0], f1 = p0[1], f2 = p1[0], f3 = p1[1];
    v16bf a;
    a[0]=(__bf16)fmaxf(f0.x,0.f); a[1]=(__bf16)fmaxf(f0.y,0.f); a[2]=(__bf16)fmaxf(f0.z,0.f); a[3]=(__bf16)fmaxf(f0.w,0.f);
    a[4]=(__bf16)fmaxf(f1.x,0.f); a[5]=(__bf16)fmaxf(f1.y,0.f); a[6]=(__bf16)fmaxf(f1.z,0.f); a[7]=(__bf16)fmaxf(f1.w,0.f);
    a[8]=(__bf16)fmaxf(f2.x,0.f); a[9]=(__bf16)fmaxf(f2.y,0.f); a[10]=(__bf16)fmaxf(f2.z,0.f); a[11]=(__bf16)fmaxf(f2.w,0.f);
    a[12]=(__bf16)fmaxf(f3.x,0.f); a[13]=(__bf16)fmaxf(f3.y,0.f); a[14]=(__bf16)fmaxf(f3.z,0.f); a[15]=(__bf16)fmaxf(f3.w,0.f);
    return a;
}

// ---------------------------------------------------------------------------
// GEMM1: h0[16 rows x 256 cols per block] = bf16(x @ W1 + b1).
// 8 waves/block, each wave owns 2 N-tiles; 16 K-steps of 32.
// ---------------------------------------------------------------------------
__global__ __launch_bounds__(256) void gemm1_kernel(
    const float* __restrict__ x, const __bf16* __restrict__ w1p,
    const float* __restrict__ b1, __bf16* __restrict__ h0)
{
    const int lane  = threadIdx.x & 31;
    const int wave  = threadIdx.x >> 5;
    const int mBase = blockIdx.x * 16;
    const int h = lane >> 4, m = lane & 15;
    const int nt0 = wave * 2, nt1 = wave * 2 + 1;
    const float* rowPtr = x + (size_t)(mBase + m) * NFEAT;

    v8f acc0 = {}; v8f acc1 = {};
    for (int kt = 0; kt < NFEAT / 32; ++kt) {
        v16bf a = load_a_frag_f32(rowPtr, kt * 32, h);
        BFrag b0, b1f;
        const v4u* bp0 = (const v4u*)(w1p + ((size_t)(kt * 16 + nt0) * 32 + lane) * 16);
        const v4u* bp1 = (const v4u*)(w1p + ((size_t)(kt * 16 + nt1) * 32 + lane) * 16);
        b0.q[0] = bp0[0]; b0.q[1] = bp0[1];
        b1f.q[0] = bp1[0]; b1f.q[1] = bp1[1];
        acc0 = __builtin_amdgcn_wmma_f32_16x16x32_bf16(false, a, false, b0.v,  (short)0, acc0, false, false);
        acc1 = __builtin_amdgcn_wmma_f32_16x16x32_bf16(false, a, false, b1f.v, (short)0, acc1, false, false);
    }
    // C/D layout: VGPR i -> row mBase + 8h + i, col = n_base + (lane&15)
    const int col0 = nt0 * 16 + m, col1 = nt1 * 16 + m;
    const float bb0 = b1[col0], bb1 = b1[col1];
#pragma unroll
    for (int i = 0; i < 8; ++i) {
        int r = mBase + 8 * h + i;
        h0[(size_t)r * NHID + col0] = (__bf16)(acc0[i] + bb0);
        h0[(size_t)r * NHID + col1] = (__bf16)(acc1[i] + bb1);
    }
}

// ---------------------------------------------------------------------------
// GEMM2: l0 = bf16(relu(hacc) @ W2 + b2), N padded 48; 3 waves/block.
// ---------------------------------------------------------------------------
__global__ __launch_bounds__(96) void gemm2_kernel(
    const float* __restrict__ hacc, const __bf16* __restrict__ w2p,
    const float* __restrict__ b2, __bf16* __restrict__ l0)
{
    const int lane  = threadIdx.x & 31;
    const int nt    = threadIdx.x >> 5;      // 0..2
    const int mBase = blockIdx.x * 16;
    const int h = lane >> 4, m = lane & 15;
    const float* rowPtr = hacc + (size_t)(mBase + m) * NHID;

    v8f acc = {};
    for (int kt = 0; kt < NHID / 32; ++kt) {
        v16bf a = load_a_frag_relu(rowPtr, kt * 32, h);
        BFrag b;
        const v4u* bp = (const v4u*)(w2p + ((size_t)(kt * 3 + nt) * 32 + lane) * 16);
        b.q[0] = bp[0]; b.q[1] = bp[1];
        acc = __builtin_amdgcn_wmma_f32_16x16x32_bf16(false, a, false, b.v, (short)0, acc, false, false);
    }
    const int col = nt * 16 + m;
    if (col < NCLASS) {
        const float bb = b2[col];
#pragma unroll
        for (int i = 0; i < 8; ++i) {
            int r = mBase + 8 * h + i;
            l0[(size_t)r * NCLASS + col] = (__bf16)(acc[i] + bb);
        }
    }
}

// ---------------------------------------------------------------------------
// SpMM1: hacc[dst][:] += w * h0[src][:]  (256 feats, one wave per edge).
// h0 read as packed bf16 pairs (u32): coalesced b32 gathers, bitop decode
// (bf16 == top half of f32, no v_cvt needed).
// ---------------------------------------------------------------------------
__global__ __launch_bounds__(256) void spmm1_kernel(
    const int* __restrict__ rows, const int* __restrict__ cols,
    const float* __restrict__ ew, const unsigned int* __restrict__ h0u,
    float* __restrict__ hacc, int E)
{
    int e = blockIdx.x * 8 + (threadIdx.x >> 5);
    if (e >= E) return;
    int lane = threadIdx.x & 31;
    int dst = rows[e], src = cols[e];
    float w = ew[e];
    const unsigned int* hp = h0u + (size_t)src * (NHID / 2);
    float* op = hacc + (size_t)dst * NHID;
#pragma unroll
    for (int j = 0; j < NHID / 64; ++j) {      // 4 pair-loads per lane
        int p = j * 32 + lane;                 // pair index: feats 2p, 2p+1
        unsigned int u = hp[p];
        float lo = __uint_as_float(u << 16);
        float hi = __uint_as_float(u & 0xffff0000u);
        atomicAdd(op + 2 * p,     w * lo);
        atomicAdd(op + 2 * p + 1, w * hi);
    }
}

// SpMM2: logits[dst][:] += w * l0[src][:]  (40 feats = 20 u32 pairs per edge)
__global__ __launch_bounds__(256) void spmm2_kernel(
    const int* __restrict__ rows, const int* __restrict__ cols,
    const float* __restrict__ ew, const unsigned int* __restrict__ l0u,
    float* __restrict__ logits, int E)
{
    int e = blockIdx.x * 8 + (threadIdx.x >> 5);
    if (e >= E) return;
    int lane = threadIdx.x & 31;
    int dst = rows[e], src = cols[e];
    float w = ew[e];
    if (lane < NCLASS / 2) {
        // l0 row stride = 40 bf16 = 20 u32 (80B, 4B-aligned per row)
        unsigned int u = l0u[(size_t)src * (NCLASS / 2) + lane];
        float lo = __uint_as_float(u << 16);
        float hi = __uint_as_float(u & 0xffff0000u);
        float* op = logits + (size_t)dst * NCLASS + 2 * lane;
        atomicAdd(op,     w * lo);
        atomicAdd(op + 1, w * hi);
    }
}

// ---------------------------------------------------------------------------
// log_softmax over 40 classes, one wave per row (wave32 shuffles).
// ---------------------------------------------------------------------------
__global__ __launch_bounds__(256) void lsm_kernel(
    const float* __restrict__ logits, float* __restrict__ out, int Nrows)
{
    int row = blockIdx.x * 8 + (threadIdx.x >> 5);
    if (row >= Nrows) return;
    int lane = threadIdx.x & 31;
    const float* lp = logits + (size_t)row * NCLASS;
    float v0 = lp[lane];
    float v1 = (lane < NCLASS - 32) ? lp[32 + lane] : -__builtin_inff();
    float mx = fmaxf(v0, v1);
#pragma unroll
    for (int o = 16; o >= 1; o >>= 1) mx = fmaxf(mx, __shfl_xor(mx, o, 32));
    float s = expf(v0 - mx) + ((lane < NCLASS - 32) ? expf(v1 - mx) : 0.0f);
#pragma unroll
    for (int o = 16; o >= 1; o >>= 1) s += __shfl_xor(s, o, 32);
    float lse = mx + logf(s);
    out[(size_t)row * NCLASS + lane] = v0 - lse;
    if (lane < NCLASS - 32)
        out[(size_t)row * NCLASS + 32 + lane] = v1 - lse;
}

__global__ void zero_f32_kernel(float* __restrict__ p, size_t n) {
    size_t i = (size_t)blockIdx.x * blockDim.x + threadIdx.x;
    size_t stride = (size_t)gridDim.x * blockDim.x;
    for (; i < n; i += stride) p[i] = 0.0f;
}

// ---------------------------------------------------------------------------
extern "C" void kernel_launch(void* const* d_in, const int* in_sizes, int n_in,
                              void* d_out, int out_size, void* d_ws, size_t ws_size,
                              hipStream_t stream) {
    const float* x  = (const float*)d_in[0];
    const int*   ei = (const int*)d_in[1];      // [2, E] (harness integer convention)
    const float* ew = (const float*)d_in[2];
    const float* W1 = (const float*)d_in[3];
    const float* b1 = (const float*)d_in[4];
    const float* W2 = (const float*)d_in[5];
    const float* b2 = (const float*)d_in[6];
    float* out = (float*)d_out;

    const int E      = in_sizes[2];             // edge_weight count
    const int Nnodes = in_sizes[0] / NFEAT;     // 100000 (divisible by 16)
    const int* rows = ei;
    const int* cols = ei + E;

    // Workspace layout (bytes, 16B aligned):
    char* ws = (char*)d_ws;
    __bf16* h0     = (__bf16*)(ws);                                    // N*256 bf16 = 51.2MB
    float*  hacc   = (float*) (ws + (size_t)Nnodes * NHID * 2);        // N*256 f32  = 102.4MB
    __bf16* l0     = (__bf16*)(ws + (size_t)Nnodes * NHID * 6);        // N*40 bf16
    float*  logits = (float*) (ws + (size_t)Nnodes * NHID * 6
                                  + (size_t)Nnodes * NCLASS * 2);      // N*40 f32
    char* tail = ws + (size_t)Nnodes * NHID * 6 + (size_t)Nnodes * NCLASS * 6;
    __bf16* w1p = (__bf16*)tail;                                       // 131072 bf16
    __bf16* w2p = (__bf16*)(tail + 131072 * 2);                        // 12288 bf16

    const int mTiles = Nnodes / 16;
    const int eBlocks = (E + 7) / 8;

    zero_f32_kernel<<<2048, 256, 0, stream>>>(hacc,   (size_t)Nnodes * NHID);
    zero_f32_kernel<<<1024, 256, 0, stream>>>(logits, (size_t)Nnodes * NCLASS);
    pack_w1_kernel<<<256, 256, 0, stream>>>(W1, w1p);
    pack_w2_kernel<<<24, 256, 0, stream>>>(W2, w2p);

    gemm1_kernel<<<mTiles, 256, 0, stream>>>(x, w1p, b1, h0);
    spmm1_kernel<<<eBlocks, 256, 0, stream>>>(rows, cols, ew,
                                              (const unsigned int*)h0, hacc, E);
    gemm2_kernel<<<mTiles, 96, 0, stream>>>(hacc, w2p, b2, l0);
    spmm2_kernel<<<eBlocks, 256, 0, stream>>>(rows, cols, ew,
                                              (const unsigned int*)l0, logits, E);
    lsm_kernel<<<(Nnodes + 7) / 8, 256, 0, stream>>>(logits, out, Nnodes);
}
